// CanineAttention_58265526338227
// MI455X (gfx1250) — matrix operand order repaired and checked by reading, not compiled, not run on hardware
//
#include <hip/hip_runtime.h>

typedef __bf16 v16bf __attribute__((ext_vector_type(16)));
typedef __bf16 v8bf  __attribute__((ext_vector_type(8)));
typedef float  v8f   __attribute__((ext_vector_type(8)));

union BF16x16 { v16bf v; v8bf h[2]; };

__device__ __forceinline__ v8f zero_v8f() {
    v8f z;
    #pragma unroll
    for (int i = 0; i < 8; ++i) z[i] = 0.0f;
    return z;
}

// pack 4 floats -> 4 bf16 and store 8 bytes to LDS
__device__ __forceinline__ void cvt_store4(__bf16* dst, float a, float b, float c, float d) {
    union { __bf16 b[4]; uint2 u; } t;
    t.b[0] = (__bf16)a; t.b[1] = (__bf16)b; t.b[2] = (__bf16)c; t.b[3] = (__bf16)d;
    *(uint2*)dst = t.u;
}

// async 16B tile copy: global (saddr + 32-bit voffset) -> LDS, tracked by ASYNCcnt
__device__ __forceinline__ void async_copy_b128(unsigned lds_addr, unsigned voff,
                                                unsigned long long gbase) {
    asm volatile("global_load_async_to_lds_b128 %0, %1, %2"
                 :: "v"(lds_addr), "v"(voff), "s"(gbase) : "memory");
}
__device__ __forceinline__ void wait_asynccnt0() {
    asm volatile("s_wait_asynccnt 0x0" ::: "memory");
}

// ---------------------------------------------------------------------------
// GEMM: out[16384,768] = X[16384,768] * W[768,768] + bias (+ residual)
// 128x128 block tile, 256 threads = 8 waves, wave owns 2(M) x 4(N) 16x16 tiles.
// Double-buffered LDS: next k-tile's global loads overlap current WMMAs.
// ---------------------------------------------------------------------------
template <typename InT, typename OutT>
__global__ __launch_bounds__(256)
void gemm_bias(const InT* __restrict__ X, const float* __restrict__ W,
               const float* __restrict__ bias, const float* __restrict__ residual,
               OutT* __restrict__ out)
{
    constexpr int LDT = 40;                    // 32 + 8 pad, keeps 16B alignment
    __shared__ __bf16 As[2][128 * LDT];
    __shared__ __bf16 Bs[2][128 * LDT];

    const int tid  = threadIdx.x;
    const int lane = tid & 31;
    const int wave = tid >> 5;
    const int mb   = blockIdx.x * 128;
    const int nb   = blockIdx.y * 128;
    const int wm   = (wave & 3) * 32;
    const int wn   = (wave >> 2) * 64;
    const int lm   = lane & 15;
    const int lh   = lane >> 4;

    v8f acc[2][4];
    #pragma unroll
    for (int i = 0; i < 2; ++i)
        #pragma unroll
        for (int j = 0; j < 4; ++j) acc[i][j] = zero_v8f();

    auto stage = [&](int buf, int k0) {
        // A tile [128 x 32]: 1024 4-element chunks, vector global load + b64 LDS store
        #pragma unroll
        for (int j = 0; j < 4; ++j) {
            const int q = j * 256 + tid;
            const int r = q >> 3, cc = (q & 7) * 4;
            const InT* src = X + (size_t)(mb + r) * 768 + k0 + cc;
            if constexpr (sizeof(InT) == 4) {
                const float4 v = *(const float4*)src;
                cvt_store4(&As[buf][r * LDT + cc], v.x, v.y, v.z, v.w);
            } else {
                *(uint2*)&As[buf][r * LDT + cc] = *(const uint2*)src;   // bf16 passthrough
            }
        }
        // B tile [32 x 128] transposed to Bs[n][k]: 4 k-values packed per b64 store
        #pragma unroll
        for (int j = 0; j < 4; ++j) {
            const int q = j * 256 + tid;
            const int n = q & 127, k = (q >> 7) * 4;
            const float* wp = W + (size_t)(k0 + k) * 768 + nb + n;
            cvt_store4(&Bs[buf][n * LDT + k], wp[0], wp[768], wp[1536], wp[2304]);
        }
    };

    stage(0, 0);
    constexpr int NK = 768 / 32;               // 24 k-tiles
    for (int kt = 0; kt < NK; ++kt) {
        __syncthreads();
        if (kt + 1 < NK) stage((kt + 1) & 1, (kt + 1) * 32);   // overlaps WMMAs below

        const __bf16* Ab = As[kt & 1];
        const __bf16* Bb = Bs[kt & 1];
        BF16x16 a[2], b[4];
        #pragma unroll
        for (int i = 0; i < 2; ++i) {
            const int row = wm + i * 16 + lm;
            a[i].h[0] = *(const v8bf*)&Ab[row * LDT + lh * 8];
            a[i].h[1] = *(const v8bf*)&Ab[row * LDT + 16 + lh * 8];
        }
        #pragma unroll
        for (int j = 0; j < 4; ++j) {
            const int n = wn + j * 16 + lm;
            b[j].h[0] = *(const v8bf*)&Bb[n * LDT + lh * 8];
            b[j].h[1] = *(const v8bf*)&Bb[n * LDT + 16 + lh * 8];
        }
        #pragma unroll
        for (int i = 0; i < 2; ++i)
            #pragma unroll
            for (int j = 0; j < 4; ++j)
                acc[i][j] = __builtin_amdgcn_wmma_f32_16x16x32_bf16(
                    false, a[i].v, false, b[j].v, (short)0, acc[i][j], false, false);
    }

    #pragma unroll
    for (int i = 0; i < 2; ++i)
        #pragma unroll
        for (int j = 0; j < 4; ++j) {
            const int col = nb + wn + j * 16 + lm;
            #pragma unroll
            for (int g = 0; g < 8; ++g) {
                const int row = mb + wm + i * 16 + lh * 8 + g;
                float v = acc[i][j][g] + bias[col];
                if (residual) v += residual[(size_t)row * 768 + col];
                out[(size_t)row * 768 + col] = (OutT)v;
            }
        }
}

// ---------------------------------------------------------------------------
// Block-diagonal attention; one block per (chunk, head, batch), 8 waves.
// Q/K tiles staged via async global->LDS b128 copies (ASYNCcnt); V transposed
// manually. Softmax in the WMMA C/D register layout via 16-lane shfl_xor.
// ---------------------------------------------------------------------------
__global__ __launch_bounds__(256)
void attn_chunk(const __bf16* __restrict__ Q, const __bf16* __restrict__ K,
                const __bf16* __restrict__ V, const float* __restrict__ mask,
                __bf16* __restrict__ ctx)
{
    const int c = blockIdx.x, h = blockIdx.y, b = blockIdx.z;
    const int tid = threadIdx.x, lane = tid & 31, wave = tid >> 5;
    const int lm = lane & 15, lh = lane >> 4;

    __shared__ __align__(16) unsigned char smem[49152];
    __bf16* Qs = (__bf16*)smem;                 // [128][64]
    __bf16* Ks = (__bf16*)(smem + 16384);       // [128][64]
    __bf16* Vt = (__bf16*)(smem + 32768);       // [64][128]  (d-major)
    __bf16* Ps = (__bf16*)smem;                 // [128][128] overlays Qs+Ks

    const size_t rowbase = (size_t)(b * 16 + c) * 128;
    const __bf16* Qg = Q + rowbase * 768 + h * 64;
    const __bf16* Kg = K + rowbase * 768 + h * 64;
    const __bf16* Vg = V + rowbase * 768 + h * 64;

    // Q/K: 1024 16B chunks each, async copy straight into LDS
    {
        const unsigned qs0 = (unsigned)(size_t)Qs;
        const unsigned ks0 = (unsigned)(size_t)Ks;
        #pragma unroll
        for (int j = 0; j < 4; ++j) {
            const int q = j * 256 + tid;
            const int r = q >> 3, d8 = (q & 7) * 8;
            const unsigned voff = (unsigned)((r * 768 + d8) * 2);
            const unsigned loff = (unsigned)((r * 64 + d8) * 2);
            async_copy_b128(qs0 + loff, voff, (unsigned long long)(size_t)Qg);
            async_copy_b128(ks0 + loff, voff, (unsigned long long)(size_t)Kg);
        }
    }
    // V: uint4 global load, transpose-scatter (2-byte stores; lanes 8 apart write adjacent r)
    #pragma unroll
    for (int j = 0; j < 4; ++j) {
        const int q = j * 256 + tid;
        const int r = q >> 3, d8 = (q & 7) * 8;
        union { uint4 u; __bf16 e[8]; } t;
        t.u = *(const uint4*)(Vg + (size_t)r * 768 + d8);
        #pragma unroll
        for (int i = 0; i < 8; ++i) Vt[(d8 + i) * 128 + r] = t.e[i];
    }
    wait_asynccnt0();
    __syncthreads();

    // S = Q K^T  (per-wave 16 rows x 128 cols, K=64 -> 2 wmma steps)
    v8f s[8];
    #pragma unroll
    for (int nt = 0; nt < 8; ++nt) s[nt] = zero_v8f();
    #pragma unroll
    for (int kt = 0; kt < 2; ++kt) {
        BF16x16 a, kb[8];
        const int qrow = wave * 16 + lm;
        a.h[0] = *(const v8bf*)&Qs[qrow * 64 + kt * 32 + lh * 8];
        a.h[1] = *(const v8bf*)&Qs[qrow * 64 + kt * 32 + 16 + lh * 8];
        #pragma unroll
        for (int nt = 0; nt < 8; ++nt) {
            const int krow = nt * 16 + lm;
            kb[nt].h[0] = *(const v8bf*)&Ks[krow * 64 + kt * 32 + lh * 8];
            kb[nt].h[1] = *(const v8bf*)&Ks[krow * 64 + kt * 32 + 16 + lh * 8];
        }
        #pragma unroll
        for (int nt = 0; nt < 8; ++nt)
            s[nt] = __builtin_amdgcn_wmma_f32_16x16x32_bf16(
                false, a.v, false, kb[nt].v, (short)0, s[nt], false, false);
    }
    __syncthreads();  // all waves done reading Qs/Ks before Ps overlays them

    // scale + mask (row q lives in vgpr g, lane half lh per the C/D layout)
    const float* mrow = mask + ((size_t)b * 2048 + (size_t)c * 128) * 2048 + c * 128;
    #pragma unroll
    for (int nt = 0; nt < 8; ++nt)
        #pragma unroll
        for (int g = 0; g < 8; ++g) {
            const int qq = wave * 16 + lh * 8 + g;
            const float m = mrow[(size_t)qq * 2048 + nt * 16 + lm];
            s[nt][g] = s[nt][g] * 0.125f + (1.0f - m) * -3.402823466e38f;
        }

    // row softmax: reduce over nt in-lane, then across the 16-lane half
    #pragma unroll
    for (int g = 0; g < 8; ++g) {
        float mx = s[0][g];
        #pragma unroll
        for (int nt = 1; nt < 8; ++nt) mx = fmaxf(mx, s[nt][g]);
        #pragma unroll
        for (int off = 1; off < 16; off <<= 1) mx = fmaxf(mx, __shfl_xor(mx, off, 32));
        float sum = 0.0f;
        #pragma unroll
        for (int nt = 0; nt < 8; ++nt) { s[nt][g] = __expf(s[nt][g] - mx); sum += s[nt][g]; }
        #pragma unroll
        for (int off = 1; off < 16; off <<= 1) sum += __shfl_xor(sum, off, 32);
        const float inv = 1.0f / sum;
        #pragma unroll
        for (int nt = 0; nt < 8; ++nt)
            Ps[(wave * 16 + lh * 8 + g) * 128 + nt * 16 + lm] = (__bf16)(s[nt][g] * inv);
    }
    __syncthreads();

    // ctx = P V  (16 rows per wave x 64 cols, K=128 -> 4 wmma steps)
    v8f o[4];
    #pragma unroll
    for (int nt = 0; nt < 4; ++nt) o[nt] = zero_v8f();
    #pragma unroll
    for (int kt = 0; kt < 4; ++kt) {
        BF16x16 pa, vb[4];
        const int prow = wave * 16 + lm;
        pa.h[0] = *(const v8bf*)&Ps[prow * 128 + kt * 32 + lh * 8];
        pa.h[1] = *(const v8bf*)&Ps[prow * 128 + kt * 32 + 16 + lh * 8];
        #pragma unroll
        for (int nt = 0; nt < 4; ++nt) {
            const int drow = nt * 16 + lm;
            vb[nt].h[0] = *(const v8bf*)&Vt[drow * 128 + kt * 32 + lh * 8];
            vb[nt].h[1] = *(const v8bf*)&Vt[drow * 128 + kt * 32 + 16 + lh * 8];
        }
        #pragma unroll
        for (int nt = 0; nt < 4; ++nt)
            o[nt] = __builtin_amdgcn_wmma_f32_16x16x32_bf16(
                false, pa.v, false, vb[nt].v, (short)0, o[nt], false, false);
    }
    #pragma unroll
    for (int nt = 0; nt < 4; ++nt)
        #pragma unroll
        for (int g = 0; g < 8; ++g) {
            const size_t row = rowbase + wave * 16 + lh * 8 + g;
            ctx[row * 768 + h * 64 + nt * 16 + lm] = (__bf16)o[nt][g];
        }
}

// ---------------------------------------------------------------------------
// LayerNorm over 768 columns; one block per row, 256 threads (3 cols each).
// ---------------------------------------------------------------------------
__global__ __launch_bounds__(256)
void layernorm_row(const float* __restrict__ X, const float* __restrict__ gamma,
                   const float* __restrict__ beta, float* __restrict__ out)
{
    const int row = blockIdx.x, tid = threadIdx.x;
    const int lane = tid & 31, wave = tid >> 5;
    const float* xr = X + (size_t)row * 768;
    const float x0 = xr[tid], x1 = xr[tid + 256], x2 = xr[tid + 512];
    float s  = x0 + x1 + x2;
    float ss = x0 * x0 + x1 * x1 + x2 * x2;
    #pragma unroll
    for (int off = 1; off < 32; off <<= 1) {
        s  += __shfl_xor(s, off, 32);
        ss += __shfl_xor(ss, off, 32);
    }
    __shared__ float red[16];
    if (lane == 0) { red[wave] = s; red[8 + wave] = ss; }
    __syncthreads();
    float ts = 0.0f, tss = 0.0f;
    #pragma unroll
    for (int w = 0; w < 8; ++w) { ts += red[w]; tss += red[8 + w]; }
    const float mu = ts * (1.0f / 768.0f);
    const float var = tss * (1.0f / 768.0f) - mu * mu;
    const float r = rsqrtf(var + 1e-12f);
    float* orow = out + (size_t)row * 768;
    orow[tid]       = (x0 - mu) * r * gamma[tid]       + beta[tid];
    orow[tid + 256] = (x1 - mu) * r * gamma[tid + 256] + beta[tid + 256];
    orow[tid + 512] = (x2 - mu) * r * gamma[tid + 512] + beta[tid + 512];
}

// ---------------------------------------------------------------------------
extern "C" void kernel_launch(void* const* d_in, const int* in_sizes, int n_in,
                              void* d_out, int out_size, void* d_ws, size_t ws_size,
                              hipStream_t stream)
{
    (void)in_sizes; (void)n_in; (void)out_size; (void)ws_size;
    const float* hidden = (const float*)d_in[0];
    const float* mask   = (const float*)d_in[1];
    const float* Wq = (const float*)d_in[2];  const float* bq = (const float*)d_in[3];
    const float* Wk = (const float*)d_in[4];  const float* bk = (const float*)d_in[5];
    const float* Wv = (const float*)d_in[6];  const float* bv = (const float*)d_in[7];
    const float* Wo = (const float*)d_in[8];  const float* bo = (const float*)d_in[9];
    const float* gamma = (const float*)d_in[10];
    const float* beta  = (const float*)d_in[11];
    float* out = (float*)d_out;

    const size_t MN = (size_t)16384 * 768;
    __bf16* Qb  = (__bf16*)d_ws;
    __bf16* Kb  = Qb + MN;
    __bf16* Vb  = Kb + MN;
    __bf16* Ctx = Vb + MN;
    float*  Pre = (float*)(Ctx + MN);

    const dim3 gemm_grid(16384 / 128, 768 / 128);
    const dim3 blk(256);

    gemm_bias<float, __bf16><<<gemm_grid, blk, 0, stream>>>(hidden, Wq, bq, nullptr, Qb);
    gemm_bias<float, __bf16><<<gemm_grid, blk, 0, stream>>>(hidden, Wk, bk, nullptr, Kb);
    gemm_bias<float, __bf16><<<gemm_grid, blk, 0, stream>>>(hidden, Wv, bv, nullptr, Vb);

    attn_chunk<<<dim3(16, 12, 8), blk, 0, stream>>>(Qb, Kb, Vb, mask, Ctx);

    gemm_bias<__bf16, float><<<gemm_grid, blk, 0, stream>>>(Ctx, Wo, bo, hidden, Pre);

    layernorm_row<<<16384, blk, 0, stream>>>(Pre, gamma, beta, out);
}
